// FastFFlayer_38740605010038
// MI455X (gfx1250) — matrix operand order repaired and checked by reading, not compiled, not run on hardware
//
#include <hip/hip_runtime.h>
#include <stdint.h>

typedef unsigned short u16;
typedef __attribute__((ext_vector_type(16))) __bf16 v16bf;
typedef __attribute__((ext_vector_type(8)))  float  v8f;

union Frag { v16bf v; uint4 q[2]; };
union U8x16 { uint4 q; u16 u[8]; };

#define SLEN 1024
#define BSZ 32
#define IN_DIM 512
#define NH 8
#define HD 64
#define NCOL 193                  // 3*64+1 per head
#define NSLOW (NH * NCOL)         // 1544
#define NPAD 1600                 // 25 * 64, padded projection width
#define ROWS (SLEN * BSZ)         // 32768

__device__ inline u16 f2bf(float f) {
  unsigned u = __float_as_uint(f);
  unsigned r = u + 0x7FFFu + ((u >> 16) & 1u);   // round-to-nearest-even
  return (u16)(r >> 16);
}
__device__ inline float bf2f(u16 b) {
  return __uint_as_float(((unsigned)b) << 16);
}

// ---------------- w_out: f32 -> bf16 ----------------
__global__ void cvt_f32_bf16(const float* __restrict__ in, u16* __restrict__ out, int n) {
  int i = blockIdx.x * blockDim.x + threadIdx.x;
  if (i < n) out[i] = f2bf(in[i]);
}

// ---------------- w_slow: f32 [K][1544] -> bf16 [K][1600] zero-padded ----------------
__global__ void cvt_pad_bf16(const float* __restrict__ in, u16* __restrict__ out) {
  int i = blockIdx.x * blockDim.x + threadIdx.x;   // over K*NPAD
  if (i >= IN_DIM * NPAD) return;
  int kk = i / NPAD, n = i % NPAD;
  out[i] = (n < NSLOW) ? f2bf(in[(size_t)kk * NSLOW + n]) : (u16)0;
}

// ---------------- LayerNorm (wave per row of 512) ----------------
__global__ __launch_bounds__(256)
void ln_kernel(const float* __restrict__ x, const float* __restrict__ gamma,
               const float* __restrict__ beta, u16* __restrict__ h) {
  int wave = threadIdx.x >> 5, lane = threadIdx.x & 31;
  int row = blockIdx.x * 8 + wave;
  const float* xr = x + (size_t)row * IN_DIM;
  float v[16];
  float s = 0.f;
  #pragma unroll
  for (int e = 0; e < 16; ++e) { v[e] = xr[lane + 32 * e]; s += v[e]; }
  #pragma unroll
  for (int m = 16; m >= 1; m >>= 1) s += __shfl_xor(s, m, 32);
  float mu = s * (1.0f / IN_DIM);
  float ss = 0.f;
  #pragma unroll
  for (int e = 0; e < 16; ++e) { float d = v[e] - mu; ss += d * d; }
  #pragma unroll
  for (int m = 16; m >= 1; m >>= 1) ss += __shfl_xor(ss, m, 32);
  float rstd = rsqrtf(ss * (1.0f / IN_DIM) + 1e-5f);
  u16* hr = h + (size_t)row * IN_DIM;
  #pragma unroll
  for (int e = 0; e < 16; ++e) {
    int c = lane + 32 * e;
    hr[c] = f2bf((v[e] - mu) * rstd * gamma[c] + beta[c]);
  }
}

// ---------------- bf16 WMMA GEMM (no edge guards; N multiple of 64) ----------------
// 256 threads, 8 waves 4(M)x2(N); wave -> 32x32 via 2x2 wmma 16x16x32.
// Register-prefetch pipeline: commit regs->LDS, barrier, prefetch next tile, compute.
#define BMT 128
#define BNT 64
#define BKT 32
#define AP 40
#define BP 40

template <bool RESID>
__global__ __launch_bounds__(256)
void gemm_bf16(const u16* __restrict__ A, const u16* __restrict__ B,
               int M, int N, int K,
               u16* __restrict__ Cbf, float* __restrict__ Cf,
               const float* __restrict__ resid) {
  __shared__ __align__(16) u16 As[BMT * AP];
  __shared__ __align__(16) u16 Bs[BNT * BP];

  int t = threadIdx.x;
  int m0 = blockIdx.y * BMT;
  int n0 = blockIdx.x * BNT;
  int lane = t & 31, g = lane >> 4, ln = lane & 15;
  int wave = t >> 5, wm = wave >> 1, wn = wave & 1;

  v8f acc[2][2];
  #pragma unroll
  for (int i = 0; i < 2; ++i)
    #pragma unroll
    for (int j = 0; j < 2; ++j)
      #pragma unroll
      for (int e = 0; e < 8; ++e) acc[i][j][e] = 0.f;

  int arow = t >> 1, aseg = (t & 1) << 4;   // A: 16 bf16 per thread
  int kb = t >> 3, nb = (t & 7) << 3;       // B: 8 bf16 per thread

  const u16* Aptr = A + (size_t)(m0 + arow) * K + aseg;
  const u16* Bptr = B + (size_t)kb * N + n0 + nb;

  uint4 pa0 = ((const uint4*)Aptr)[0];
  uint4 pa1 = ((const uint4*)Aptr)[1];
  uint4 pb  = *(const uint4*)Bptr;

  for (int kk = 0; kk < K; kk += BKT) {
    // ---- commit staged registers into LDS
    uint4* dA = (uint4*)&As[arow * AP + aseg];
    dA[0] = pa0;
    dA[1] = pa1;
    U8x16 tb; tb.q = pb;
    #pragma unroll
    for (int j = 0; j < 8; ++j) Bs[(nb + j) * BP + kb] = tb.u[j];
    __syncthreads();

    // ---- prefetch next K-tile (latency hidden under WMMA below)
    if (kk + BKT < K) {
      pa0 = ((const uint4*)(Aptr + kk + BKT))[0];
      pa1 = ((const uint4*)(Aptr + kk + BKT))[1];
      pb  = *(const uint4*)(Bptr + (size_t)(kk + BKT) * N);
    }

    // ---- fragments (ISA 7.12.2 16-bit layouts): two b128 LDS loads each
    Frag af[2], bfr[2];
    #pragma unroll
    for (int mt = 0; mt < 2; ++mt) {
      int r = wm * 32 + mt * 16 + ln;
      af[mt].q[0] = *(const uint4*)&As[r * AP + 8 * g];
      af[mt].q[1] = *(const uint4*)&As[r * AP + 16 + 8 * g];
    }
    #pragma unroll
    for (int nt = 0; nt < 2; ++nt) {
      int n = wn * 32 + nt * 16 + ln;
      bfr[nt].q[0] = *(const uint4*)&Bs[n * BP + 8 * g];
      bfr[nt].q[1] = *(const uint4*)&Bs[n * BP + 16 + 8 * g];
    }
    #pragma unroll
    for (int mt = 0; mt < 2; ++mt)
      #pragma unroll
      for (int nt = 0; nt < 2; ++nt)
        acc[mt][nt] = __builtin_amdgcn_wmma_f32_16x16x32_bf16(
            false, af[mt].v, false, bfr[nt].v, (short)0, acc[mt][nt], false, false);

    __syncthreads();
  }

  // ---- epilogue: C/D layout VGPR r -> M = r + 8g, N = ln
  #pragma unroll
  for (int mt = 0; mt < 2; ++mt) {
    #pragma unroll
    for (int nt = 0; nt < 2; ++nt) {
      #pragma unroll
      for (int r = 0; r < 8; ++r) {
        int row = m0 + wm * 32 + mt * 16 + g * 8 + r;
        int col = n0 + wn * 32 + nt * 16 + ln;
        size_t idx = (size_t)row * N + col;
        if (RESID) {
          Cf[idx] = acc[mt][nt][r] + resid[idx];
        } else {
          Cbf[idx] = f2bf(acc[mt][nt][r]);
        }
      }
    }
  }
}

// ---------------- activations: beta sigmoid, q/k = sumnorm(elu+1) ----------------
__global__ __launch_bounds__(256)
void act_kernel(const u16* __restrict__ qkvb,
                u16* __restrict__ q, u16* __restrict__ k,
                u16* __restrict__ v, float* __restrict__ beta) {
  int r = blockIdx.x;
  int hh = threadIdx.x >> 5, lane = threadIdx.x & 31;
  size_t base = (size_t)r * NPAD + hh * NCOL;
  size_t ob = ((size_t)r * NH + hh) * HD;

  float a0 = bf2f(qkvb[base + lane]);
  float a1 = bf2f(qkvb[base + 32 + lane]);
  a0 = a0 > 0.f ? a0 + 1.f : expf(a0);
  a1 = a1 > 0.f ? a1 + 1.f : expf(a1);
  float s = a0 + a1;
  #pragma unroll
  for (int m = 16; m >= 1; m >>= 1) s += __shfl_xor(s, m, 32);
  float inv = 1.f / s;
  q[ob + lane] = f2bf(a0 * inv);
  q[ob + 32 + lane] = f2bf(a1 * inv);

  float b0 = bf2f(qkvb[base + 64 + lane]);
  float b1 = bf2f(qkvb[base + 96 + lane]);
  b0 = b0 > 0.f ? b0 + 1.f : expf(b0);
  b1 = b1 > 0.f ? b1 + 1.f : expf(b1);
  float sk = b0 + b1;
  #pragma unroll
  for (int m = 16; m >= 1; m >>= 1) sk += __shfl_xor(sk, m, 32);
  float invk = 1.f / sk;
  k[ob + lane] = f2bf(b0 * invk);
  k[ob + 32 + lane] = f2bf(b1 * invk);

  v[ob + lane] = qkvb[base + 128 + lane];
  v[ob + 32 + lane] = qkvb[base + 160 + lane];

  if (lane == 0)
    beta[(size_t)r * NH + hh] = 1.f / (1.f + expf(-bf2f(qkvb[base + 192])));
}

// ---------------- async stage: one 16B chunk of {k,v,q} into LDS ----------------
__device__ __forceinline__ void issue_async(const u16* kp, const u16* vp, const u16* qp,
                                            size_t base, int t, uint32_t ldst) {
  const u16* gp = (t < 8 ? kp : (t < 16 ? vp : qp)) + base + (size_t)(t & 7) * 8;
  asm volatile("global_load_async_to_lds_b128 %0, %1, off"
               :: "v"(ldst), "v"(gp) : "memory");
}

// ---------------- delta-rule recurrence: block per (b,h), W in VGPRs ----------------
__global__ __launch_bounds__(512)
void delta_kernel(const u16* __restrict__ q, const u16* __restrict__ k,
                  const u16* __restrict__ v, const float* __restrict__ beta,
                  u16* __restrict__ o) {
  int bh = blockIdx.x;
  int b = bh >> 3, hh = bh & 7;
  int t = threadIdx.x;
  int i = t & 63, g = t >> 6;        // row i, col-group g (8 cols each)

  float w[8];
  #pragma unroll
  for (int j = 0; j < 8; ++j) w[j] = 0.f;

  // double-buffered raw bf16 staging: [buf][ k:0..63 | v:64..127 | q:128..191 ]
  __shared__ __align__(16) u16 kvq[2][192];
  __shared__ float ub[64];
  __shared__ float red[8][64];

  // LDS aperture keeps the byte offset in the low 32 bits of the flat address
  uint32_t lbase = (uint32_t)(uintptr_t)(&kvq[0][0]);
  uint32_t lchunk = (uint32_t)((t >> 3) * 128 + (t & 7) * 16);

  // prologue: stage ts=0 into buffer 0
  if (t < 24)
    issue_async(k, v, q, ((size_t)0 * BSZ + b) * NH * HD + (size_t)hh * HD, t, lbase + lchunk);
  asm volatile("s_wait_asynccnt 0x0" ::: "memory");
  __syncthreads();

  for (int ts = 0; ts < SLEN; ++ts) {
    int buf = ts & 1;
    size_t base = (((size_t)ts * BSZ + b) * NH + hh) * HD;
    // prefetch ts+1 into the other buffer (safe: readers of it finished last iter)
    if (ts + 1 < SLEN && t < 24) {
      size_t nbase = (((size_t)(ts + 1) * BSZ + b) * NH + hh) * HD;
      issue_async(k, v, q, nbase, t, lbase + (buf ^ 1) * 384u + lchunk);
    }
    float sb = beta[((size_t)ts * BSZ + b) * NH + hh];

    // k chunk for this thread's 8 columns (one b128 LDS load)
    U8x16 kr; kr.q = *(const uint4*)&kvq[buf][g * 8];
    float kf[8];
    #pragma unroll
    for (int j = 0; j < 8; ++j) kf[j] = bf2f(kr.u[j]);

    // v_old = W @ k
    float p = 0.f;
    #pragma unroll
    for (int j = 0; j < 8; ++j) p += w[j] * kf[j];
    red[g][i] = p;
    __syncthreads();
    if (t < 64) {
      float vo = 0.f;
      #pragma unroll
      for (int gg = 0; gg < 8; ++gg) vo += red[gg][i];
      float vt = bf2f(kvq[buf][64 + i]);
      ub[i] = sb * (vt - vo);
    }
    __syncthreads();

    // W += ub * k^T ; o = W @ q (fused)
    float u = ub[i];
    U8x16 qr; qr.q = *(const uint4*)&kvq[buf][128 + g * 8];
    p = 0.f;
    #pragma unroll
    for (int j = 0; j < 8; ++j) {
      w[j] += u * kf[j];
      p += w[j] * bf2f(qr.u[j]);
    }
    red[g][i] = p;
    __syncthreads();
    if (t < 64) {
      float oo = 0.f;
      #pragma unroll
      for (int gg = 0; gg < 8; ++gg) oo += red[gg][i];
      o[base + i] = f2bf(oo);
    }
    // publish next buffer: issuing wave drains ASYNCcnt, then all waves barrier
    asm volatile("s_wait_asynccnt 0x0" ::: "memory");
    __syncthreads();
  }
}

// ---------------- launcher ----------------
extern "C" void kernel_launch(void* const* d_in, const int* in_sizes, int n_in,
                              void* d_out, int out_size, void* d_ws, size_t ws_size,
                              hipStream_t stream) {
  const float* x      = (const float*)d_in[0];
  const float* gamma  = (const float*)d_in[1];
  const float* lbeta  = (const float*)d_in[2];
  const float* w_slow = (const float*)d_in[3];
  const float* w_out  = (const float*)d_in[4];
  float* out = (float*)d_out;

  char* ws = (char*)d_ws;
  u16* h        = (u16*)ws;  ws += (size_t)ROWS * IN_DIM * 2;
  u16* wslow_bf = (u16*)ws;  ws += (size_t)IN_DIM * NPAD * 2;
  u16* wout_bf  = (u16*)ws;  ws += (size_t)IN_DIM * IN_DIM * 2;
  u16* qkvb     = (u16*)ws;  ws += (size_t)ROWS * NPAD * 2;
  u16* qb       = (u16*)ws;  ws += (size_t)ROWS * IN_DIM * 2;
  u16* kb       = (u16*)ws;  ws += (size_t)ROWS * IN_DIM * 2;
  u16* vb       = (u16*)ws;  ws += (size_t)ROWS * IN_DIM * 2;
  float* betab  = (float*)ws; ws += (size_t)ROWS * NH * 4;
  u16* ob       = (u16*)ws;  ws += (size_t)ROWS * IN_DIM * 2;

  int nws = IN_DIM * NPAD;
  cvt_pad_bf16<<<(nws + 255) / 256, 256, 0, stream>>>(w_slow, wslow_bf);
  int nwo = IN_DIM * IN_DIM;
  cvt_f32_bf16<<<(nwo + 255) / 256, 256, 0, stream>>>(w_out, wout_bf, nwo);

  ln_kernel<<<ROWS / 8, 256, 0, stream>>>(x, gamma, lbeta, h);

  gemm_bf16<false><<<dim3(NPAD / BNT, ROWS / BMT), 256, 0, stream>>>(
      h, wslow_bf, ROWS, NPAD, IN_DIM, qkvb, (float*)nullptr, (const float*)nullptr);

  act_kernel<<<ROWS, 256, 0, stream>>>(qkvb, qb, kb, vb, betab);

  delta_kernel<<<BSZ * NH, 512, 0, stream>>>(qb, kb, vb, betab, ob);

  gemm_bf16<true><<<dim3(IN_DIM / BNT, ROWS / BMT), 256, 0, stream>>>(
      ob, wout_bf, ROWS, IN_DIM, IN_DIM, (u16*)nullptr, out, x);
}
